// CrossBlock_88502096101921
// MI455X (gfx1250) — matrix operand order repaired and checked by reading, not compiled, not run on hardware
//
#include <hip/hip_runtime.h>
#include <hip/hip_bf16.h>

typedef __attribute__((ext_vector_type(16))) _Float16 v16h;
typedef __attribute__((ext_vector_type(8)))  _Float16 v8h;
typedef __attribute__((ext_vector_type(8)))  float    v8f;

#define EMBED 256
#define DH    64
#define HEADS 4
#define NTOK  2048
#define BATCH 4
#define MROWS (BATCH * NTOK)   // 8192 token rows per stream

// ---------------------------------------------------------------------------
// Fragment helpers (wave32 WMMA; K order below is this code's contiguous choice, not the ISA table order)
// A 16x32 f16: lane holds row=lane%16; K run = 16*(lane/16)+i
//              ->  two 16B loads at p0 and p0+8 halves.
// B 32x16 f16: lane holds col=lane%16; K run = 16*(lane/16)+i -> 32B contiguous.
// C/D 16x16 f32: lane holds col=lane%16, c[r] = row r + 8*(lane/16).
// ---------------------------------------------------------------------------
__device__ __forceinline__ v8f zero8() {
    v8f z;
#pragma unroll
    for (int i = 0; i < 8; ++i) z[i] = 0.f;
    return z;
}

__device__ __forceinline__ v16h ldA(const _Float16* p0) {
    v16h a;
    ((v8h*)&a)[0] = *(const v8h*)(p0);
    ((v8h*)&a)[1] = *(const v8h*)(p0 + 8);
    return a;
}

__device__ __forceinline__ v16h ldB(const _Float16* p0) {
    v16h b;
    ((v8h*)&b)[0] = *(const v8h*)(p0);
    ((v8h*)&b)[1] = *(const v8h*)(p0 + 8);
    return b;
}

__device__ __forceinline__ v8f wmma32(v16h a, v16h b, v8f c) {
    return __builtin_amdgcn_wmma_f32_16x16x32_f16(false, a, false, b,
                                                  (short)0, c, false, false);
}

// ---------------------------------------------------------------------------
// Weight transpose + f32->f16 convert:  W[k][n] (f32) -> Wt[n][k] (f16)
// ---------------------------------------------------------------------------
__global__ __launch_bounds__(256)
void transpose_w(const float* __restrict__ W, _Float16* __restrict__ Wt,
                 int K, int N) {
    int idx = blockIdx.x * 256 + threadIdx.x;
    if (idx >= K * N) return;
    int k = idx / N, n = idx - k * N;
    Wt[(long)n * K + k] = (_Float16)W[idx];
}

// x [M,256] f32  ->  cat [M,512] f16 (columns 0..255)
__global__ __launch_bounds__(256)
void convert_x(const float* __restrict__ x, _Float16* __restrict__ cat) {
    int idx = blockIdx.x * 256 + threadIdx.x;   // < MROWS*EMBED
    int row = idx >> 8, col = idx & 255;
    cat[(long)row * 512 + col] = (_Float16)x[idx];
}

// ---------------------------------------------------------------------------
// Generic GEMM:  C = epilogue( A[M,K](f16,lda) @ Bt[N][K](f16) )
//   val = (acc + bias[col]) * scale  (+ resid[row*N+col] if RESID)
// Software-pipelined: fragments for k0+32 are prefetched into distinct
// registers while the 4 WMMAs for k0 issue, so global loads stay in flight.
// grid = (M/128, N/64); 256 threads = 8 waves, wave -> 16x64 tile.
// ---------------------------------------------------------------------------
template <bool OUTF16, bool RESID>
__global__ __launch_bounds__(256)
void gemm_f16(const _Float16* __restrict__ A, int lda,
              const _Float16* __restrict__ Bt,
              const float* __restrict__ bias, float scale,
              float* __restrict__ Cf32, _Float16* __restrict__ Cf16,
              int ldc, int col0, const float* __restrict__ resid,
              int N, int K) {
    const int lane = threadIdx.x & 31;
    const int wave = threadIdx.x >> 5;
    const int row0 = blockIdx.x * 128 + wave * 16;
    const int n0   = blockIdx.y * 64;
    const int l16  = lane & 15;
    const int half = lane >> 4;

    v8f c[4];
#pragma unroll
    for (int g = 0; g < 4; ++g) c[g] = zero8();

    const _Float16* Ap = A + (long)(row0 + l16) * lda + half * 16;
    const _Float16* Bp[4];
#pragma unroll
    for (int g = 0; g < 4; ++g)
        Bp[g] = Bt + (long)(n0 + g * 16 + l16) * K + half * 16;

    // prologue: fragments for k0 = 0
    v16h a0 = ldA(Ap);
    v16h b0[4];
#pragma unroll
    for (int g = 0; g < 4; ++g) b0[g] = ldB(Bp[g]);

    int k0 = 0;
    for (; k0 + 32 < K; k0 += 32) {
        // prefetch next k-chunk into fresh registers
        v16h a1 = ldA(Ap + k0 + 32);
        v16h b1[4];
#pragma unroll
        for (int g = 0; g < 4; ++g) b1[g] = ldB(Bp[g] + k0 + 32);
        // consume current chunk
#pragma unroll
        for (int g = 0; g < 4; ++g) c[g] = wmma32(a0, b0[g], c[g]);
        a0 = a1;
#pragma unroll
        for (int g = 0; g < 4; ++g) b0[g] = b1[g];
    }
#pragma unroll
    for (int g = 0; g < 4; ++g) c[g] = wmma32(a0, b0[g], c[g]);

#pragma unroll
    for (int g = 0; g < 4; ++g) {
        int col = n0 + g * 16 + l16;
        float bz = bias[col];
#pragma unroll
        for (int r = 0; r < 8; ++r) {
            int row = row0 + r + 8 * half;
            float v = (c[g][r] + bz) * scale;
            if (RESID) v += resid[(long)row * N + col];
            long idx = (long)row * ldc + col0 + col;
            if (OUTF16) Cf16[idx] = (_Float16)v;
            else        Cf32[idx] = v;
        }
    }
}

// ---------------------------------------------------------------------------
// Flash cross-attention: Om[row, h*64+d] = softmax_j(Q_row . K_j) @ V
// Q,K,V: [B*N, 256] f16 (head h = columns h*64..h*64+63), qk pre-scaled.
// grid = (N/128, HEADS, BATCH); 8 waves x 16 rows; KV tiles of 32 in LDS.
// Next tile's global loads are issued before the compute phase so HBM/L2
// latency hides behind the 8 WMMAs + softmax of the current tile.
// ---------------------------------------------------------------------------
__global__ __launch_bounds__(256)
void attn_flash(const _Float16* __restrict__ Q,
                const _Float16* __restrict__ Kg,
                const _Float16* __restrict__ Vg,
                _Float16* __restrict__ Om) {
    __shared__ _Float16 Kt[32][64];      // K tile, row-major [j][d]
    __shared__ _Float16 Vt[64][32];      // V tile transposed [d][j]
    __shared__ _Float16 Pb[8][16][32];   // per-wave P staging

    const int tid  = threadIdx.x;
    const int lane = tid & 31, wave = tid >> 5;
    const int l16  = lane & 15, half = lane >> 4;
    const int h = blockIdx.y, b = blockIdx.z;
    const int rowbase = b * NTOK + blockIdx.x * 128 + wave * 16;
    const int hofs = h * DH;

    // Q A-fragments (K-chunks 0..31, 32..63 of the 64-wide head)
    const _Float16* qp = Q + (long)(rowbase + l16) * EMBED + hofs + half * 16;
    v16h aq0 = ldA(qp);
    v16h aq1 = ldA(qp + 32);

    v8f acc[4];
#pragma unroll
    for (int g = 0; g < 4; ++g) acc[g] = zero8();
    float mrow[8], lrow[8];
#pragma unroll
    for (int r = 0; r < 8; ++r) { mrow[r] = -1e30f; lrow[r] = 0.f; }

    const int jt = tid >> 3;         // 0..31
    const int dt = (tid & 7) * 8;    // 0..56
    const long tbase = (long)(b * NTOK + jt) * EMBED + hofs + dt;

    // prefetch tile 0
    v8h kreg = *(const v8h*)(Kg + tbase);
    v8h vreg = *(const v8h*)(Vg + tbase);

    for (int j0 = 0; j0 < NTOK; j0 += 32) {
        __syncthreads();                     // prior tile fully consumed
        // deposit current tile into LDS
        *(v8h*)&Kt[jt][dt] = kreg;
#pragma unroll
        for (int i = 0; i < 8; ++i) Vt[dt + i][jt] = vreg[i];
        // issue next tile's global loads (wraps on last iter; harmless)
        {
            const long gn = tbase + (long)((j0 + 32) & (NTOK - 1)) * EMBED;
            kreg = *(const v8h*)(Kg + gn);
            vreg = *(const v8h*)(Vg + gn);
        }
        __syncthreads();

        // S = Q (16x64) @ K^T (64x32): two 16-col groups x two K-chunks
        v8f s0 = zero8(), s1 = zero8();
        s0 = wmma32(aq0, ldB(&Kt[l16][half * 16]), s0);
        s0 = wmma32(aq1, ldB(&Kt[l16][32 + half * 16]), s0);
        s1 = wmma32(aq0, ldB(&Kt[16 + l16][half * 16]), s1);
        s1 = wmma32(aq1, ldB(&Kt[16 + l16][32 + half * 16]), s1);

        // online softmax per row (row = r + 8*half; cols striped over 16 lanes)
#pragma unroll
        for (int r = 0; r < 8; ++r) {
            float mx = fmaxf(s0[r], s1[r]);
            mx = fmaxf(mx, __shfl_xor(mx, 1));
            mx = fmaxf(mx, __shfl_xor(mx, 2));
            mx = fmaxf(mx, __shfl_xor(mx, 4));
            mx = fmaxf(mx, __shfl_xor(mx, 8));
            float mnew = fmaxf(mrow[r], mx);
            float corr = __expf(mrow[r] - mnew);
            float p0 = __expf(s0[r] - mnew);
            float p1 = __expf(s1[r] - mnew);
            float ps = p0 + p1;
            ps += __shfl_xor(ps, 1);
            ps += __shfl_xor(ps, 2);
            ps += __shfl_xor(ps, 4);
            ps += __shfl_xor(ps, 8);
            lrow[r] = lrow[r] * corr + ps;
            mrow[r] = mnew;
            s0[r] = p0; s1[r] = p1;
#pragma unroll
            for (int g = 0; g < 4; ++g) acc[g][r] *= corr;
        }

        // stage P (16x32) to LDS in C-layout, re-read in A-layout
#pragma unroll
        for (int r = 0; r < 8; ++r) {
            Pb[wave][r + 8 * half][l16]      = (_Float16)s0[r];
            Pb[wave][r + 8 * half][16 + l16] = (_Float16)s1[r];
        }
        asm volatile("s_wait_dscnt 0" ::: "memory");  // same-wave LDS RAW fence
        v16h ap = ldA(&Pb[wave][l16][half * 16]);

        // acc += P (16x32) @ V (32x64), output d-chunks of 16
#pragma unroll
        for (int g = 0; g < 4; ++g)
            acc[g] = wmma32(ap, ldB(&Vt[g * 16 + l16][half * 16]), acc[g]);
    }

    // write merged head output [row][h*64 + d] as f16
#pragma unroll
    for (int g = 0; g < 4; ++g) {
#pragma unroll
        for (int r = 0; r < 8; ++r) {
            int row = rowbase + r + 8 * half;
            float v = acc[g][r] / lrow[r];
            Om[(long)row * EMBED + hofs + g * 16 + l16] = (_Float16)v;
        }
    }
}

// ---------------------------------------------------------------------------
// LayerNorm(512) + exact GELU, one wave per row; writes f16.
// ---------------------------------------------------------------------------
__global__ __launch_bounds__(256)
void ln_gelu(const float* __restrict__ H, const float* __restrict__ gma,
             const float* __restrict__ bta, _Float16* __restrict__ out) {
    const int lane = threadIdx.x & 31;
    const long row = (long)blockIdx.x * 8 + (threadIdx.x >> 5);
    const float* hp = H + row * 512;
    float vals[16], s = 0.f, s2 = 0.f;
#pragma unroll
    for (int i = 0; i < 16; ++i) {
        float v = hp[lane + i * 32];
        vals[i] = v; s += v; s2 += v * v;
    }
#pragma unroll
    for (int m = 1; m < 32; m <<= 1) {
        s  += __shfl_xor(s,  m);
        s2 += __shfl_xor(s2, m);
    }
    float mu   = s * (1.f / 512.f);
    float var  = s2 * (1.f / 512.f) - mu * mu;
    float rstd = rsqrtf(var + 1e-5f);
#pragma unroll
    for (int i = 0; i < 16; ++i) {
        int col = lane + i * 32;
        float v = (vals[i] - mu) * rstd * gma[col] + bta[col];
        float g = 0.5f * v * (1.f + erff(v * 0.7071067811865476f));
        out[row * 512 + col] = (_Float16)g;
    }
}

// ---------------------------------------------------------------------------
extern "C" void kernel_launch(void* const* d_in, const int* in_sizes, int n_in,
                              void* d_out, int out_size, void* d_ws, size_t ws_size,
                              hipStream_t stream) {
    const float* x0   = (const float*)d_in[0];
    const float* x1   = (const float*)d_in[1];
    const float* Wqk  = (const float*)d_in[2];
    const float* bqk  = (const float*)d_in[3];
    const float* Wv   = (const float*)d_in[4];
    const float* bv   = (const float*)d_in[5];
    const float* Wo   = (const float*)d_in[6];
    const float* bo   = (const float*)d_in[7];
    const float* W1   = (const float*)d_in[8];
    const float* b1   = (const float*)d_in[9];
    const float* ln_g = (const float*)d_in[10];
    const float* ln_b = (const float*)d_in[11];
    const float* W2   = (const float*)d_in[12];
    const float* b2   = (const float*)d_in[13];
    float* out = (float*)d_out;

    // workspace carve-out
    char* w = (char*)d_ws;
    auto alloc = [&](size_t bytes) -> char* {
        char* p = w; w += (bytes + 255) & ~(size_t)255; return p;
    };
    _Float16* cat0 = (_Float16*)alloc((size_t)MROWS * 512 * 2); // x||m, reused as GELU out
    _Float16* cat1 = (_Float16*)alloc((size_t)MROWS * 512 * 2);
    _Float16* qk0  = (_Float16*)alloc((size_t)MROWS * EMBED * 2);
    _Float16* qk1  = (_Float16*)alloc((size_t)MROWS * EMBED * 2);
    _Float16* v0   = (_Float16*)alloc((size_t)MROWS * EMBED * 2);
    _Float16* v1   = (_Float16*)alloc((size_t)MROWS * EMBED * 2);
    _Float16* m0m  = (_Float16*)alloc((size_t)MROWS * EMBED * 2);
    _Float16* m1m  = (_Float16*)alloc((size_t)MROWS * EMBED * 2);
    float*    h0   = (float*)alloc((size_t)MROWS * 512 * 4);
    float*    h1   = (float*)alloc((size_t)MROWS * 512 * 4);
    _Float16* Wqkt = (_Float16*)alloc((size_t)256 * 256 * 2);
    _Float16* Wvt  = (_Float16*)alloc((size_t)256 * 256 * 2);
    _Float16* Wot  = (_Float16*)alloc((size_t)256 * 256 * 2);
    _Float16* W1t  = (_Float16*)alloc((size_t)512 * 512 * 2);
    _Float16* W2t  = (_Float16*)alloc((size_t)512 * 256 * 2);

    // 1. weight transposes (f32 -> f16, [k][n] -> [n][k])
    transpose_w<<<(256 * 256) / 256, 256, 0, stream>>>(Wqk, Wqkt, 256, 256);
    transpose_w<<<(256 * 256) / 256, 256, 0, stream>>>(Wv,  Wvt,  256, 256);
    transpose_w<<<(256 * 256) / 256, 256, 0, stream>>>(Wo,  Wot,  256, 256);
    transpose_w<<<(512 * 512) / 256, 256, 0, stream>>>(W1,  W1t,  512, 512);
    transpose_w<<<(512 * 256) / 256, 256, 0, stream>>>(W2,  W2t,  512, 256);

    // 2. x -> f16 into concat buffers (cols 0..255)
    convert_x<<<(MROWS * EMBED) / 256, 256, 0, stream>>>(x0, cat0);
    convert_x<<<(MROWS * EMBED) / 256, 256, 0, stream>>>(x1, cat1);

    const float qsc = 0.35355339059327376f;  // DH^-0.25 (SCALE**0.5)
    dim3 gP(MROWS / 128, EMBED / 64);

    // 3. projections (A = concat buffer, lda=512, first 256 cols; K=256)
    gemm_f16<true,false><<<gP, 256, 0, stream>>>(cat0, 512, Wqkt, bqk, qsc, nullptr, qk0, 256, 0, nullptr, 256, 256);
    gemm_f16<true,false><<<gP, 256, 0, stream>>>(cat1, 512, Wqkt, bqk, qsc, nullptr, qk1, 256, 0, nullptr, 256, 256);
    gemm_f16<true,false><<<gP, 256, 0, stream>>>(cat0, 512, Wvt,  bv,  1.f, nullptr, v0,  256, 0, nullptr, 256, 256);
    gemm_f16<true,false><<<gP, 256, 0, stream>>>(cat1, 512, Wvt,  bv,  1.f, nullptr, v1,  256, 0, nullptr, 256, 256);

    // 4. flash cross-attention (row softmax of sim, and of sim^T)
    dim3 gA(NTOK / 128, HEADS, BATCH);
    attn_flash<<<gA, 256, 0, stream>>>(qk0, qk1, v1, m0m);
    attn_flash<<<gA, 256, 0, stream>>>(qk1, qk0, v0, m1m);

    // 5. output projection -> concat cols 256..511
    gemm_f16<true,false><<<gP, 256, 0, stream>>>(m0m, 256, Wot, bo, 1.f, nullptr, cat0, 512, 256, nullptr, 256, 256);
    gemm_f16<true,false><<<gP, 256, 0, stream>>>(m1m, 256, Wot, bo, 1.f, nullptr, cat1, 512, 256, nullptr, 256, 256);

    // 6. FFN layer 1 (512 -> 512, f32 out)
    dim3 g1(MROWS / 128, 512 / 64);
    gemm_f16<false,false><<<g1, 256, 0, stream>>>(cat0, 512, W1t, b1, 1.f, h0, nullptr, 512, 0, nullptr, 512, 512);
    gemm_f16<false,false><<<g1, 256, 0, stream>>>(cat1, 512, W1t, b1, 1.f, h1, nullptr, 512, 0, nullptr, 512, 512);

    // 7. LayerNorm + GELU (overwrites concat buffers with f16 activations)
    ln_gelu<<<MROWS / 8, 256, 0, stream>>>(h0, ln_g, ln_b, cat0);
    ln_gelu<<<MROWS / 8, 256, 0, stream>>>(h1, ln_g, ln_b, cat1);

    // 8. FFN layer 2 (512 -> 256) + residual, straight to d_out
    gemm_f16<false,true><<<gP, 256, 0, stream>>>(cat0, 512, W2t, b2, 1.f, out, nullptr, 256, 0, x0, 256, 512);
    gemm_f16<false,true><<<gP, 256, 0, stream>>>(cat1, 512, W2t, b2, 1.f,
                                     out + (size_t)MROWS * EMBED, nullptr, 256, 0, x1, 256, 512);
}
